// TopologyFeatures_50457275793852
// MI455X (gfx1250) — 16-bit and 8-bit matrix operand order repaired and checked by reading, compiled, not run on hardware
//
#include <hip/hip_runtime.h>
#include <hip/hip_bf16.h>
#include <math.h>


#define N_ASSETS 512
#define WINDOW   256
#define BATCH    32
#define N_SQ     9

typedef __attribute__((ext_vector_type(8)))  _Float16 v8h;
typedef __attribute__((ext_vector_type(16))) _Float16 v16h;
typedef __attribute__((ext_vector_type(8)))  float    v8f;
typedef __attribute__((ext_vector_type(4)))  int      v4i;
typedef __attribute__((ext_vector_type(8)))  int      v8i;

union AHalf { v16h v; v8h h[2]; };
union AByte { v8i v; unsigned long long q[4]; };
union BByte { v8i v; v4i o[2]; };

__global__ void topo_normalize_k(const float* __restrict__ ret,
                                 _Float16* __restrict__ xnT) {
    int idx = blockIdx.x * blockDim.x + threadIdx.x;
    int b = idx / N_ASSETS;
    int i = idx - b * N_ASSETS;
    const float* p = ret + (size_t)b * WINDOW * N_ASSETS + i;
    float s = 0.f, ss = 0.f;
    for (int w = 0; w < WINDOW; ++w) {
        float x = p[(size_t)w * N_ASSETS];
        s += x; ss += x * x;
    }
    float mean = s * (1.0f / WINDOW);
    float var  = (ss - (float)WINDOW * mean * mean) * (1.0f / (WINDOW - 1));
    float inv  = 1.0f / (sqrtf(fmaxf(var, 0.0f)) + 1e-8f);
    _Float16* q = xnT + ((size_t)b * N_ASSETS + i) * WINDOW;
    for (int w = 0; w < WINDOW; ++w)
        q[w] = (_Float16)((p[(size_t)w * N_ASSETS] - mean) * inv);
}

__global__ void topo_corr_k(const _Float16* __restrict__ xnT,
                            float* __restrict__ corr) {
    int wave = (blockIdx.x * blockDim.x + threadIdx.x) >> 5;
    int lane = threadIdx.x & 31;
    int tj = wave & 31;
    int ti = (wave >> 5) & 31;
    int b  = wave >> 10;
    int m  = lane & 15;
    int hi = lane >> 4;

    const _Float16* pA = xnT + ((size_t)(b * N_ASSETS + ti * 16 + m)) * WINDOW;
    const _Float16* pB = xnT + ((size_t)(b * N_ASSETS + tj * 16 + m)) * WINDOW;

    v8f acc = {};
#pragma unroll
    for (int k = 0; k < WINDOW; k += 32) {
        AHalf a;
        a.h[0] = *(const v8h*)(pA + k + 16 * hi);
        a.h[1] = *(const v8h*)(pA + k + 16 * hi + 8);
        v16h bv = *(const v16h*)(pB + k + 16 * hi);
        acc = __builtin_amdgcn_wmma_f32_16x16x32_f16(
            false, a.v, false, bv, (short)0, acc, false, false);
    }

    const float scale = 1.0f / WINDOW;
    float* outb = corr + (size_t)b * N_ASSETS * N_ASSETS;
#pragma unroll
    for (int r = 0; r < 8; ++r)
        outb[(size_t)(ti * 16 + r + 8 * hi) * N_ASSETS + tj * 16 + m] = acc[r] * scale;
}

__global__ void topo_adj_k(const float* __restrict__ corr,
                           unsigned char* __restrict__ R, float t) {
    size_t e = ((size_t)blockIdx.x * blockDim.x + threadIdx.x) * 4;
    int rowflat = (int)(e / N_ASSETS);
    int i  = rowflat & (N_ASSETS - 1);
    int j0 = (int)(e & (N_ASSETS - 1));
    const float* c = corr + e;
    unsigned pack = 0;
#pragma unroll
    for (int k = 0; k < 4; ++k) {
        int j = j0 + k;
        unsigned v = (i == j) ? 1u : (fabsf(c[k]) > t ? 1u : 0u);
        pack |= v << (8 * k);
    }
    *(unsigned*)(R + e) = pack;
}

__global__ void topo_closure_k(const unsigned char* __restrict__ Rin,
                               unsigned char* __restrict__ Rout) {
    int wave = (blockIdx.x * blockDim.x + threadIdx.x) >> 5;
    int lane = threadIdx.x & 31;
    int tj = wave & 15;
    int ti = (wave >> 4) & 15;
    int b  = wave >> 8;
    int m  = lane & 15;
    int hi = lane >> 4;

    const unsigned char* base = Rin + (size_t)b * N_ASSETS * N_ASSETS;
    const unsigned char* pA0 = base + (size_t)(ti * 32 + m)      * N_ASSETS;
    const unsigned char* pA1 = base + (size_t)(ti * 32 + 16 + m) * N_ASSETS;
    const unsigned char* pB0 = base + (size_t)(tj * 32 + m)      * N_ASSETS;
    const unsigned char* pB1 = base + (size_t)(tj * 32 + 16 + m) * N_ASSETS;

    v8i acc00 = {}, acc01 = {}, acc10 = {}, acc11 = {};
#pragma unroll
    for (int k = 0; k < N_ASSETS; k += 64) {
        AByte a0, a1;
#pragma unroll
        for (int g = 0; g < 4; ++g) {
            a0.q[g] = *(const unsigned long long*)(pA0 + k + 16 * hi + 8 * (g & 1) + 32 * (g >> 1));
            a1.q[g] = *(const unsigned long long*)(pA1 + k + 16 * hi + 8 * (g & 1) + 32 * (g >> 1));
        }
        BByte b0, b1;
        b0.o[0] = *(const v4i*)(pB0 + k + 16 * hi);
        b0.o[1] = *(const v4i*)(pB0 + k + 32 + 16 * hi);
        b1.o[0] = *(const v4i*)(pB1 + k + 16 * hi);
        b1.o[1] = *(const v4i*)(pB1 + k + 32 + 16 * hi);
        if (k + 64 < N_ASSETS) {
            __builtin_prefetch(pA0 + k + 64, 0, 1);
            __builtin_prefetch(pA1 + k + 64, 0, 1);
            __builtin_prefetch(pB0 + k + 64, 0, 1);
            __builtin_prefetch(pB1 + k + 64, 0, 1);
        }
        acc00 = __builtin_amdgcn_wmma_i32_16x16x64_iu8(
            false, a0.v, false, b0.v, acc00, false, false);
        acc01 = __builtin_amdgcn_wmma_i32_16x16x64_iu8(
            false, a0.v, false, b1.v, acc01, false, false);
        acc10 = __builtin_amdgcn_wmma_i32_16x16x64_iu8(
            false, a1.v, false, b0.v, acc10, false, false);
        acc11 = __builtin_amdgcn_wmma_i32_16x16x64_iu8(
            false, a1.v, false, b1.v, acc11, false, false);
    }

    unsigned char* pO = Rout + (size_t)b * N_ASSETS * N_ASSETS;
    int r0 = ti * 32 + 8 * hi;
    int c0 = tj * 32 + m;
#pragma unroll
    for (int r = 0; r < 8; ++r) {
        pO[(size_t)(r0 + r)      * N_ASSETS + c0]      = (acc00[r] > 0) ? 1 : 0;
        pO[(size_t)(r0 + r)      * N_ASSETS + c0 + 16] = (acc01[r] > 0) ? 1 : 0;
        pO[(size_t)(r0 + 16 + r) * N_ASSETS + c0]      = (acc10[r] > 0) ? 1 : 0;
        pO[(size_t)(r0 + 16 + r) * N_ASSETS + c0 + 16] = (acc11[r] > 0) ? 1 : 0;
    }
}

__global__ void topo_feat_k(const unsigned char* __restrict__ Rfin,
                            const float* __restrict__ corr,
                            float t, int ti, float* __restrict__ out) {
    int b = blockIdx.x;
    __shared__ int s_comp, s_edges;
    if (threadIdx.x == 0) { s_comp = 0; s_edges = 0; }
    __syncthreads();

    int comp = 0, edges = 0;
    for (int i = threadIdx.x; i < N_ASSETS; i += blockDim.x) {
        const unsigned char* row = Rfin + ((size_t)b * N_ASSETS + i) * N_ASSETS;
        int self = 1;
        for (int j = 0; j < i; ++j)
            if (row[j]) { self = 0; break; }
        comp += self;
        const float* crow = corr + ((size_t)b * N_ASSETS + i) * N_ASSETS;
        for (int j = 0; j < N_ASSETS; ++j)
            if (j != i && fabsf(crow[j]) > t) ++edges;
    }
    atomicAdd(&s_comp, comp);
    atomicAdd(&s_edges, edges);
    __syncthreads();

    if (threadIdx.x == 0) {
        float c = (float)s_comp;
        float e = (float)s_edges * 0.5f;
        out[b * 6 + 2 * ti]     = c / (float)N_ASSETS;
        out[b * 6 + 2 * ti + 1] = fmaxf(0.0f, e - (float)N_ASSETS + c) / (float)N_ASSETS;
    }
}

extern "C" void kernel_launch(void* const* d_in, const int* in_sizes, int n_in,
                              void* d_out, int out_size, void* d_ws, size_t ws_size,
                              hipStream_t stream) {
    (void)in_sizes; (void)n_in; (void)out_size; (void)ws_size;
    const float* returns = (const float*)d_in[0];
    float* out = (float*)d_out;

    char* ws = (char*)d_ws;
    const size_t xnT_bytes  = (size_t)BATCH * N_ASSETS * WINDOW * 2;
    const size_t corr_bytes = (size_t)BATCH * N_ASSETS * N_ASSETS * 4;
    const size_t R_bytes    = (size_t)BATCH * N_ASSETS * N_ASSETS;
    _Float16*      xnT  = (_Float16*)ws;
    float*         corr = (float*)(ws + xnT_bytes);
    unsigned char* R0   = (unsigned char*)(ws + xnT_bytes + corr_bytes);
    unsigned char* R1   = R0 + R_bytes;

    topo_normalize_k<<<(BATCH * N_ASSETS) / 256, 256, 0, stream>>>(returns, xnT);

    topo_corr_k<<<8192, 128, 0, stream>>>(xnT, corr);

    const float th[3] = {0.3f, 0.5f, 0.7f};
    for (int ti = 0; ti < 3; ++ti) {
        topo_adj_k<<<(int)((size_t)BATCH * N_ASSETS * N_ASSETS / 4 / 256), 256, 0,
                     stream>>>(corr, R0, th[ti]);
        unsigned char* a = R0;
        unsigned char* bptr = R1;
        for (int s = 0; s < N_SQ; ++s) {
            topo_closure_k<<<2048, 128, 0, stream>>>(a, bptr);
            unsigned char* tmp = a; a = bptr; bptr = tmp;
        }
        topo_feat_k<<<BATCH, 256, 0, stream>>>(a, corr, th[ti], ti, out);
    }
}
